// FlashBack_52381421142471
// MI455X (gfx1250) — matrix operand order repaired and checked by reading, not compiled, not run on hardware
//
#include <hip/hip_runtime.h>

#define S_LEN 2048
#define E_DIM 512
#define H_DIM 512
#define G3    1536
#define V_DIM 50000
#define WIN   11
#define GRU_BLOCKS 64

typedef __attribute__((ext_vector_type(16))) __bf16 bf16x16;
typedef __attribute__((ext_vector_type(8)))  float  f32x8;
typedef __attribute__((ext_vector_type(4)))  float  f32x4;
typedef __attribute__((ext_vector_type(4)))  __bf16 bf16x4;

union FragBF { bf16x16 v; uint4 q[2]; };
union Cvt4   { bf16x4 h; uint2 u; };

// packed f32x4 -> bf16x4 (RNE); lowers to v_cvt_pk_bf16_f32 when available
__device__ __forceinline__ uint2 cvt4(const float4 f) {
    f32x4 v; v.x = f.x; v.y = f.y; v.z = f.z; v.w = f.w;
    Cvt4 c; c.h = __builtin_convertvector(v, bf16x4);
    return c.u;
}

// ---------------------------------------------------------------------------
// Generic GEMM: C[m][n] = sum_k A[idx(m)][k] * B[n][k] + bias[n]
// fp32 inputs converted to bf16, v_wmma_f32_16x16x32_bf16 accumulate.
// Block = 256 thr (8 waves, 2x4), tile M64 x N128, BK=64, double-buffered LDS
// (one barrier per iteration; next-tile global loads overlap WMMAs).
// Each wave: 32x32 output = 4 accumulators, 8 WMMAs per iteration.
// Requires M % 64 == 0, K % 64 == 0.
// ---------------------------------------------------------------------------
#define LDS_STRIDE 72   // 64 k + 8 pad (144B rows: 16B-aligned, bank-skewed)

__global__ __launch_bounds__(256)
void gemm_bf16_wmma(const float* __restrict__ A, const int* __restrict__ Aidx,
                    const float* __restrict__ B, const float* __restrict__ bias,
                    float* __restrict__ C, int N, int K, int ldc)
{
    __shared__ __align__(16) unsigned short As[2][64  * LDS_STRIDE];  // 2 x  9 KB
    __shared__ __align__(16) unsigned short Bs[2][128 * LDS_STRIDE];  // 2 x 18 KB

    const int tid  = threadIdx.x;
    const int lane = tid & 31;
    const int wv   = tid >> 5;
    const int wr   = wv & 1;    // 2 wave-rows -> 32 M each
    const int wc   = wv >> 1;   // 4 wave-cols -> 32 N each
    const int m0   = blockIdx.y * 64;
    const int n0   = blockIdx.x * 128;

    // cooperative tile loads: A 64x64 (16 f32/thr), B 128x64 (32 f32/thr)
    const int arow = tid >> 2;
    const int akb  = (tid & 3) << 4;           // 0,16,32,48
    const int agrow = Aidx ? Aidx[m0 + arow] : (m0 + arow);
    const float* aptr = A + (size_t)agrow * K + akb;

    const int brow = tid >> 1;
    const int bkb  = (tid & 1) << 5;           // 0,32
    const int bcol = n0 + brow;
    const bool bvalid = bcol < N;
    const float* bptr = B + (size_t)bcol * K + bkb;

    const int half = lane >> 4;   // 0: lanes 0-15, 1: lanes 16-31
    const int l16  = lane & 15;

    f32x8 acc[2][2] = {};
    float4 ar[4];   // A staging: 16 f32
    float4 br[8];   // B staging: 32 f32

    // ---- prologue: fetch tile 0 ----
#pragma unroll
    for (int q = 0; q < 4; ++q) ar[q] = *(const float4*)(aptr + q * 4);
#pragma unroll
    for (int q = 0; q < 8; ++q) {
        if (bvalid) br[q] = *(const float4*)(bptr + q * 4);
        else { br[q].x = br[q].y = br[q].z = br[q].w = 0.f; }
    }
#pragma unroll
    for (int q = 0; q < 2; ++q) {
        const uint2 c0 = cvt4(ar[2 * q]), c1 = cvt4(ar[2 * q + 1]);
        uint4 p; p.x = c0.x; p.y = c0.y; p.z = c1.x; p.w = c1.y;
        *(uint4*)&As[0][arow * LDS_STRIDE + akb + q * 8] = p;
    }
#pragma unroll
    for (int q = 0; q < 4; ++q) {
        const uint2 c0 = cvt4(br[2 * q]), c1 = cvt4(br[2 * q + 1]);
        uint4 p; p.x = c0.x; p.y = c0.y; p.z = c1.x; p.w = c1.y;
        *(uint4*)&Bs[0][brow * LDS_STRIDE + bkb + q * 8] = p;
    }
    __syncthreads();

    const int NIT = K >> 6;
    for (int it = 0; it < NIT; ++it) {
        const int cur = it & 1, nxt = cur ^ 1;

        // issue next tile's global loads (overlap with WMMAs below)
        if (it + 1 < NIT) {
            const int k0 = (it + 1) << 6;
#pragma unroll
            for (int q = 0; q < 4; ++q) ar[q] = *(const float4*)(aptr + k0 + q * 4);
#pragma unroll
            for (int q = 0; q < 8; ++q) {
                if (bvalid) br[q] = *(const float4*)(bptr + k0 + q * 4);
                else { br[q].x = br[q].y = br[q].z = br[q].w = 0.f; }
            }
        }

        // 8 WMMAs out of current buffers
        const unsigned short* Ac = As[cur];
        const unsigned short* Bc = Bs[cur];
#pragma unroll
        for (int ks = 0; ks < 2; ++ks) {
            const int ko = ks * 32;
            // A frags: lanes 0-15 -> k{0..7,16..23}; lanes 16-31 -> k{8..15,24..31}
            FragBF fa0, fa1, fb0, fb1;
            const int am = wr * 32 + l16;
            fa0.q[0] = *(const uint4*)&Ac[am * LDS_STRIDE + ko + half * 8];
            fa0.q[1] = *(const uint4*)&Ac[am * LDS_STRIDE + ko + half * 8 + 16];
            fa1.q[0] = *(const uint4*)&Ac[(am + 16) * LDS_STRIDE + ko + half * 8];
            fa1.q[1] = *(const uint4*)&Ac[(am + 16) * LDS_STRIDE + ko + half * 8 + 16];
            // B frags: lanes 0-15 -> k{0..7,16..23}; lanes 16-31 -> k{8..15,24..31}
            const int bn = wc * 32 + l16;
            fb0.q[0] = *(const uint4*)&Bc[bn * LDS_STRIDE + ko + half * 8];
            fb0.q[1] = *(const uint4*)&Bc[bn * LDS_STRIDE + ko + half * 8 + 16];
            fb1.q[0] = *(const uint4*)&Bc[(bn + 16) * LDS_STRIDE + ko + half * 8];
            fb1.q[1] = *(const uint4*)&Bc[(bn + 16) * LDS_STRIDE + ko + half * 8 + 16];

            acc[0][0] = __builtin_amdgcn_wmma_f32_16x16x32_bf16(false, fa0.v, false, fb0.v,
                                                                (short)0, acc[0][0], false, false);
            acc[0][1] = __builtin_amdgcn_wmma_f32_16x16x32_bf16(false, fa0.v, false, fb1.v,
                                                                (short)0, acc[0][1], false, false);
            acc[1][0] = __builtin_amdgcn_wmma_f32_16x16x32_bf16(false, fa1.v, false, fb0.v,
                                                                (short)0, acc[1][0], false, false);
            acc[1][1] = __builtin_amdgcn_wmma_f32_16x16x32_bf16(false, fa1.v, false, fb1.v,
                                                                (short)0, acc[1][1], false, false);
        }

        // convert + stage next tile into the other buffer
        if (it + 1 < NIT) {
#pragma unroll
            for (int q = 0; q < 2; ++q) {
                const uint2 c0 = cvt4(ar[2 * q]), c1 = cvt4(ar[2 * q + 1]);
                uint4 p; p.x = c0.x; p.y = c0.y; p.z = c1.x; p.w = c1.y;
                *(uint4*)&As[nxt][arow * LDS_STRIDE + akb + q * 8] = p;
            }
#pragma unroll
            for (int q = 0; q < 4; ++q) {
                const uint2 c0 = cvt4(br[2 * q]), c1 = cvt4(br[2 * q + 1]);
                uint4 p; p.x = c0.x; p.y = c0.y; p.z = c1.x; p.w = c1.y;
                *(uint4*)&Bs[nxt][brow * LDS_STRIDE + bkb + q * 8] = p;
            }
        }
        __syncthreads();
    }

    // C/D layout: lane -> n = l16; VGPR r -> m = r + 8*half
#pragma unroll
    for (int j = 0; j < 2; ++j) {
        const int nc = n0 + wc * 32 + j * 16 + l16;
        if (nc < N) {
            const float bi = bias[nc];
#pragma unroll
            for (int i = 0; i < 2; ++i) {
                const int mb = m0 + wr * 32 + i * 16 + half * 8;
#pragma unroll
                for (int r = 0; r < 8; ++r)
                    C[(size_t)(mb + r) * ldc + nc] = acc[i][j][r] + bi;
            }
        }
    }
}

// ---------------------------------------------------------------------------
// GRU scan: 64 persistent blocks, each owns 8 hidden units (24 W_hh rows,
// 48KB LDS-resident). Global step barrier via monotonic atomic counter,
// h double-buffered by step parity in scratch.
// ---------------------------------------------------------------------------
__global__ __launch_bounds__(256)
void gru_scan(const float* __restrict__ W_hh, const float* __restrict__ b_hh,
              const float* __restrict__ xg, float* __restrict__ hseq,
              float* __restrict__ hglob, unsigned int* __restrict__ counter)
{
    __shared__ float Wl[24 * 512];
    __shared__ float hbuf[512];
    __shared__ float hgl[24];
    __shared__ float bl[24];

    const int g   = blockIdx.x;
    const int tid = threadIdx.x;

    for (int lr = 0; lr < 24; ++lr) {
        const int gate = lr >> 3, il = lr & 7;
        const int row  = gate * H_DIM + g * 8 + il;
        for (int k = tid; k < 512; k += 256)
            Wl[lr * 512 + k] = W_hh[(size_t)row * 512 + k];
        if (tid == 0) bl[lr] = b_hh[row];
    }
    for (int k = tid; k < 512; k += 256) hbuf[k] = 0.f;
    __syncthreads();

    const int lr = tid >> 3;   // 0..31 (rows 0..23 active)
    const int sl = tid & 7;    // 8 lanes per row

    for (int t = 0; t < S_LEN; ++t) {
        if (lr < 24) {
            const float* wrow = &Wl[lr * 512 + sl * 64];
            const float* hb   = &hbuf[sl * 64];
            float s = 0.f;
#pragma unroll 8
            for (int k = 0; k < 64; ++k) s = fmaf(wrow[k], hb[k], s);
            s += __shfl_xor(s, 1, 8);
            s += __shfl_xor(s, 2, 8);
            s += __shfl_xor(s, 4, 8);
            if (sl == 0) hgl[lr] = s + bl[lr];
        }
        __syncthreads();

        if (tid < 8) {
            const int i = g * 8 + tid;
            const size_t xb = (size_t)t * G3;
            const float xr = xg[xb + i];
            const float xz = xg[xb + 512 + i];
            const float xn = xg[xb + 1024 + i];
            const float hr = hgl[tid], hz = hgl[8 + tid], hn = hgl[16 + tid];
            const float r = 1.f / (1.f + __expf(-(xr + hr)));
            const float z = 1.f / (1.f + __expf(-(xz + hz)));
            const float n = tanhf(xn + r * hn);
            const float hnew = (1.f - z) * n + z * hbuf[i];
            hglob[((t + 1) & 1) * 512 + i] = hnew;
            hseq[(size_t)t * 512 + i]     = hnew;
            __threadfence();
        }
        __syncthreads();

        if (tid == 0) {
            __hip_atomic_fetch_add(counter, 1u, __ATOMIC_RELEASE, __HIP_MEMORY_SCOPE_AGENT);
            const unsigned target = (unsigned)(GRU_BLOCKS * (t + 1));
            while (__hip_atomic_load(counter, __ATOMIC_ACQUIRE, __HIP_MEMORY_SCOPE_AGENT) < target)
                __builtin_amdgcn_s_sleep(1);
        }
        __syncthreads();
        for (int k = tid; k < 512; k += 256)
            hbuf[k] = hglob[((t + 1) & 1) * 512 + k];
        __syncthreads();
    }
}

__global__ void init_misc(unsigned int* counter) {
    if (threadIdx.x == 0) *counter = 0u;
}

// ---------------------------------------------------------------------------
// Flashback weighted window + SELU. One block per sequence position.
// ---------------------------------------------------------------------------
__global__ __launch_bounds__(256)
void flashback_selu(const float* __restrict__ tim, const float* __restrict__ coord,
                    const float* __restrict__ hseq, float* __restrict__ hsel)
{
    __shared__ float wl[16];
    __shared__ int   jcl[16];
    __shared__ float wsum;
    const int s   = blockIdx.x;
    const int tid = threadIdx.x;

    if (tid < WIN) {
        const int j  = s - tid;
        const int jc = j < 0 ? 0 : j;
        const float dt = tim[s] - tim[jc];
        const float dx = coord[2 * s]     - coord[2 * jc];
        const float dy = coord[2 * s + 1] - coord[2 * jc + 1];
        const float sq = dx * dx + dy * dy;
        const float dsv = sq > 0.f ? sqrtf(sq) : 0.f;
        const float ft = (__cosf(dt * (6.28318530717958647692f / 86400.f)) + 1.f) * 0.5f *
                         __expf(-(dt / 86400.f) * 0.1f);
        const float fs = __expf(-dsv * 50.f);
        wl[tid]  = (j >= 0) ? (ft * fs + 1e-10f) : 0.f;
        jcl[tid] = jc;
    }
    __syncthreads();
    if (tid == 0) {
        float sm = 0.f;
        for (int d = 0; d < WIN; ++d) sm += wl[d];
        wsum = sm;
    }
    __syncthreads();
    const float inv = 1.f / wsum;
    for (int k = tid; k < H_DIM; k += 256) {
        float acc = 0.f;
#pragma unroll
        for (int d = 0; d < WIN; ++d)
            acc = fmaf(wl[d], hseq[(size_t)jcl[d] * H_DIM + k], acc);
        const float x = acc * inv;
        const float alpha = 1.6732632423543772f, scale = 1.0507009873554805f;
        const float y = x > 0.f ? scale * x : scale * alpha * (__expf(x) - 1.f);
        hsel[(size_t)s * H_DIM + k] = y;
    }
}

// ---------------------------------------------------------------------------
// Per-row online logsumexp over V, then in-place normalize.
// ---------------------------------------------------------------------------
__global__ __launch_bounds__(256)
void row_lse(const float* __restrict__ y, float* __restrict__ lse, int N)
{
    __shared__ float sm[256], sa[256];
    const int s = blockIdx.x, tid = threadIdx.x;
    const float* row = y + (size_t)s * N;
    float m = -3.4e38f, a = 0.f;
    for (int n = tid; n < N; n += 256) {
        const float v = row[n];
        if (v > m) { a = a * __expf(m - v) + 1.f; m = v; }
        else       { a += __expf(v - m); }
    }
    sm[tid] = m; sa[tid] = a;
    __syncthreads();
    for (int off = 128; off > 0; off >>= 1) {
        if (tid < off) {
            const float m2 = sm[tid + off], a2 = sa[tid + off];
            const float M  = fmaxf(sm[tid], m2);
            sa[tid] = sa[tid] * __expf(sm[tid] - M) + a2 * __expf(m2 - M);
            sm[tid] = M;
        }
        __syncthreads();
    }
    if (tid == 0) lse[s] = sm[0] + __logf(sa[0]);
}

__global__ __launch_bounds__(256)
void sub_lse(float* __restrict__ y, const float* __restrict__ lse, int N)
{
    const int s = blockIdx.y;
    const int n = blockIdx.x * 256 + threadIdx.x;
    if (n < N) y[(size_t)s * N + n] -= lse[s];
}

// ---------------------------------------------------------------------------
extern "C" void kernel_launch(void* const* d_in, const int* in_sizes, int n_in,
                              void* d_out, int out_size, void* d_ws, size_t ws_size,
                              hipStream_t stream)
{
    const int*   loc   = (const int*)  d_in[0];
    const float* tim   = (const float*)d_in[1];
    const float* coord = (const float*)d_in[2];
    const float* emb   = (const float*)d_in[3];
    const float* W_ih  = (const float*)d_in[4];
    const float* W_hh  = (const float*)d_in[5];
    const float* b_ih  = (const float*)d_in[6];
    const float* b_hh  = (const float*)d_in[7];
    const float* fc_w  = (const float*)d_in[8];
    const float* fc_b  = (const float*)d_in[9];
    float* out = (float*)d_out;

    float* ws    = (float*)d_ws;
    float* xg    = ws;                            // 2048*1536
    float* hseq  = xg   + (size_t)S_LEN * G3;     // 2048*512
    float* hsel  = hseq + (size_t)S_LEN * H_DIM;  // 2048*512
    float* hglob = hsel + (size_t)S_LEN * H_DIM;  // 2*512
    float* lse   = hglob + 1024;                  // 2048
    unsigned int* counter = (unsigned int*)(lse + S_LEN);

    // 1) xg = emb[loc] @ W_ih^T + b_ih   (M=2048, N=1536, K=512)
    gemm_bf16_wmma<<<dim3(G3 / 128, S_LEN / 64), 256, 0, stream>>>(
        emb, loc, W_ih, b_ih, xg, G3, E_DIM, G3);

    // 2) GRU scan (64 cooperating persistent blocks)
    init_misc<<<1, 64, 0, stream>>>(counter);
    gru_scan<<<GRU_BLOCKS, 256, 0, stream>>>(W_hh, b_hh, xg, hseq, hglob, counter);

    // 3) flashback weighted window + SELU
    flashback_selu<<<S_LEN, 256, 0, stream>>>(tim, coord, hseq, hsel);

    // 4) y = hsel @ fc_w^T + fc_b   (M=2048, N=50000, K=512)
    gemm_bf16_wmma<<<dim3((V_DIM + 127) / 128, S_LEN / 64), 256, 0, stream>>>(
        hsel, nullptr, fc_w, fc_b, out, V_DIM, H_DIM, V_DIM);

    // 5) log_softmax
    row_lse<<<S_LEN, 256, 0, stream>>>(out, lse, V_DIM);
    sub_lse<<<dim3((V_DIM + 255) / 256, S_LEN), 256, 0, stream>>>(out, lse, V_DIM);
}
